// ASIS_70660801953946
// MI455X (gfx1250) — compile-verified, matrix operand order checked by reading, not run on hardware
//
#include <hip/hip_runtime.h>
#include <cstdint>
#include <cstddef>

// ---------------------------------------------------------------------------
// ASIS (PointNet++ sem+ins segmentation) forward pass for gfx1250 (MI455X).
// GEMMs run on v_wmma_f32_16x16x32_f16 fed from double-buffered LDS f16 tiles;
// irregular ops (FPS / ball query / 3-NN interp) are plain wave32 kernels.
// ---------------------------------------------------------------------------

typedef __attribute__((ext_vector_type(16))) _Float16 v16h;
typedef __attribute__((ext_vector_type(8)))  _Float16 v8h;
typedef __attribute__((ext_vector_type(8)))  float    v8f;

#define TPB 256

static inline int ceil_div(int a, int b) { return (a + b - 1) / b; }

// ---------------------------------------------------------------------------
// Farthest point sampling: one block per batch. mind[] lives in LDS.
// ---------------------------------------------------------------------------
__global__ void fps_kernel(const float* __restrict__ p, int* __restrict__ idx,
                           int N, int m) {
  __shared__ float mind[4096];
  __shared__ float rv[TPB];
  __shared__ int   ri[TPB];
  __shared__ int   s_last;
  const int b = blockIdx.x, tid = threadIdx.x;
  const float* pb = p + (size_t)b * N * 3;

  for (int j = tid; j < N; j += blockDim.x) mind[j] = 1e10f;
  if (tid == 0) s_last = 0;
  __syncthreads();

  for (int i = 0; i < m; ++i) {
    const int last = s_last;
    if (tid == 0) idx[b * m + i] = last;
    const float cx = pb[last * 3 + 0], cy = pb[last * 3 + 1], cz = pb[last * 3 + 2];
    float bestv = -1.0f; int bestj = 0;
    for (int j = tid; j < N; j += blockDim.x) {
      const float dx = pb[j * 3 + 0] - cx;
      const float dy = pb[j * 3 + 1] - cy;
      const float dz = pb[j * 3 + 2] - cz;
      const float d2 = dx * dx + dy * dy + dz * dz;
      const float nd = fminf(mind[j], d2);
      mind[j] = nd;
      if (nd > bestv) { bestv = nd; bestj = j; }
    }
    rv[tid] = bestv; ri[tid] = bestj;
    __syncthreads();
    for (int s = blockDim.x >> 1; s > 0; s >>= 1) {
      if (tid < s) {
        if (rv[tid + s] > rv[tid] ||
            (rv[tid + s] == rv[tid] && ri[tid + s] < ri[tid])) {
          rv[tid] = rv[tid + s]; ri[tid] = ri[tid + s];
        }
      }
      __syncthreads();
    }
    if (tid == 0) s_last = ri[0];
    __syncthreads();
  }
}

// q[b,m,:] = p[b, idx[b,m], :]
__global__ void gather_pts_k(const float* __restrict__ p, const int* __restrict__ idx,
                             float* __restrict__ q, int N, int M, int total) {
  const int t = blockIdx.x * blockDim.x + threadIdx.x;
  if (t >= total) return;
  const int b = t / M;
  const int j = idx[t];
  const float* s = p + ((size_t)b * N + j) * 3;
  float* d = q + (size_t)t * 3;
  d[0] = s[0]; d[1] = s[1]; d[2] = s[2];
}

// First-k neighbors within radius; pad with first hit (matches ref semantics).
__global__ void ball_group_k(const float* __restrict__ p, const float* __restrict__ q,
                             int* __restrict__ bidx, int N, int M, float r2, int k,
                             int total) {
  const int t = blockIdx.x * blockDim.x + threadIdx.x;
  if (t >= total) return;
  const int b = t / M;
  const float* pb = p + (size_t)b * N * 3;
  const float qx = q[(size_t)t * 3 + 0];
  const float qy = q[(size_t)t * 3 + 1];
  const float qz = q[(size_t)t * 3 + 2];
  int cnt = 0, first = 0;
  for (int j = 0; j < N; ++j) {
    const float dx = pb[j * 3 + 0] - qx;
    const float dy = pb[j * 3 + 1] - qy;
    const float dz = pb[j * 3 + 2] - qz;
    const float d2 = dx * dx + dy * dy + dz * dz;
    if (d2 <= r2) {
      if (cnt == 0) first = j;
      bidx[(size_t)t * k + cnt] = j;
      if (++cnt == k) break;
    }
  }
  for (; cnt < k; ++cnt) bidx[(size_t)t * k + cnt] = first;
}

// h[row, :] = [p[j]-q[m] (3), x[j] (C)]   row = ((b*M)+m)*k + kk
__global__ void build_h_k(const float* __restrict__ p, const float* __restrict__ x,
                          const float* __restrict__ q, const int* __restrict__ bidx,
                          float* __restrict__ h, int N, int M, int k, int C,
                          int total) {
  const int t = blockIdx.x * blockDim.x + threadIdx.x;
  if (t >= total) return;
  const int b   = t / (M * k);
  const int rem = t - b * (M * k);
  const int mm  = rem / k;
  const int j   = bidx[t];
  const float* pj = p + ((size_t)b * N + j) * 3;
  const float* qm = q + ((size_t)b * M + mm) * 3;
  float* hr = h + (size_t)t * (3 + C);
  hr[0] = pj[0] - qm[0];
  hr[1] = pj[1] - qm[1];
  hr[2] = pj[2] - qm[2];
  const float* xj = x + ((size_t)b * N + j) * C;
  for (int c = 0; c < C; ++c) hr[3 + c] = xj[c];
}

// out[rm, c] = max_kk a[(rm*k+kk), c]
__global__ void maxpool_k(const float* __restrict__ a, float* __restrict__ o,
                          int rows, int k, int C, int total) {
  const int t = blockIdx.x * blockDim.x + threadIdx.x;
  if (t >= total) return;
  const int rm = t / C, c = t - rm * C;
  float v = -1e30f;
  for (int kk = 0; kk < k; ++kk)
    v = fmaxf(v, a[((size_t)rm * k + kk) * C + c]);
  o[t] = v;
}

// 3-NN inverse-distance interpolation, writing into a strided concat dest.
__global__ void interp3_k(const float* __restrict__ pt, const float* __restrict__ pf,
                          const float* __restrict__ xf, float* __restrict__ dst,
                          int Nt, int Nf, int C, int dstStride, int dstOff,
                          int total) {
  const int t = blockIdx.x * blockDim.x + threadIdx.x;
  if (t >= total) return;
  const int b = t / Nt;
  const float qx = pt[(size_t)t * 3 + 0];
  const float qy = pt[(size_t)t * 3 + 1];
  const float qz = pt[(size_t)t * 3 + 2];
  const float* pfb = pf + (size_t)b * Nf * 3;
  float d0 = 1e30f, d1 = 1e30f, d2v = 1e30f;
  int   i0 = 0,     i1 = 0,     i2 = 0;
  for (int j = 0; j < Nf; ++j) {
    const float dx = pfb[j * 3 + 0] - qx;
    const float dy = pfb[j * 3 + 1] - qy;
    const float dz = pfb[j * 3 + 2] - qz;
    const float d = dx * dx + dy * dy + dz * dz;
    if (d < d0)       { d2v = d1; i2 = i1; d1 = d0; i1 = i0; d0 = d; i0 = j; }
    else if (d < d1)  { d2v = d1; i2 = i1; d1 = d;  i1 = j; }
    else if (d < d2v) { d2v = d;  i2 = j; }
  }
  float w0 = 1.0f / fmaxf(d0, 1e-10f);
  float w1 = 1.0f / fmaxf(d1, 1e-10f);
  float w2 = 1.0f / fmaxf(d2v, 1e-10f);
  const float s = w0 + w1 + w2;
  w0 /= s; w1 /= s; w2 /= s;
  const float* x0 = xf + ((size_t)b * Nf + i0) * C;
  const float* x1 = xf + ((size_t)b * Nf + i1) * C;
  const float* x2 = xf + ((size_t)b * Nf + i2) * C;
  float* drow = dst + (size_t)t * dstStride + dstOff;
  for (int c = 0; c < C; ++c)
    drow[c] = w0 * x0[c] + w1 * x1[c] + w2 * x2[c];
}

// Copy src[rows, Csrc] into strided concat destination at channel offset.
__global__ void copy_strided_k(const float* __restrict__ src, float* __restrict__ dst,
                               int Csrc, int dstStride, int dstOff, int total) {
  const int t = blockIdx.x * blockDim.x + threadIdx.x;
  if (t >= total) return;
  const int r = t / Csrc, c = t - r * Csrc;
  dst[(size_t)r * dstStride + dstOff + c] = src[t];
}

// ---------------------------------------------------------------------------
// WMMA GEMM: Y[M,N] = act( X[M,K] * W[K,N] + (res ? res : 0) )
//
// Block = 256 threads = 8 waves = 2x4 grid of 16x16 WMMA tiles (32x64 block
// tile). K is consumed in 32-wide chunks through DOUBLE-BUFFERED LDS panels:
// while chunk `ch` is computed from buffer ch&1, chunk `ch+1` is staged into
// the other buffer -> one barrier per chunk, global loads overlap WMMA.
// Global loads are unconditional (clamped addresses; K-tail zeroed by a
// post-load select) so no EXEC-predicated loads in the hot loop.
// Fragments: two 16B-aligned contiguous LDS runs per operand
// (K(e) = (e&7) + (e>>3)*16 + (lane>>4)*8) -> ds_load_b128 pairs.
// Epilogue: single per-lane column predicate; res/relu uniform branches are
// hoisted outside the 8-row loops so residual loads batch under one wait.
// ---------------------------------------------------------------------------
#define GEMM_BM 32
#define GEMM_BN 64
#define GEMM_BK 32

__global__ void gemm_wmma_k(const float* __restrict__ X, const float* __restrict__ W,
                            const float* __restrict__ res, float* __restrict__ Y,
                            int M, int K, int N, int relu) {
  __shared__ __align__(16) _Float16 As[2][GEMM_BM][GEMM_BK];  // [buf][m][k]
  __shared__ __align__(16) _Float16 Bs[2][GEMM_BN][GEMM_BK];  // [buf][n][k]

  const int tid  = threadIdx.x;
  const int lane = tid & 31;
  const int w    = tid >> 5;            // wave 0..7
  const int l15  = lane & 15;
  const int g    = lane >> 4;

  const int bm = blockIdx.y * GEMM_BM;
  const int bn = blockIdx.x * GEMM_BN;
  const int tm = bm + (w >> 2) * 16;    // this wave's output tile origin
  const int tn = bn + (w & 3) * 16;

  // staging coordinates (whole block cooperates)
  const int ar = tid >> 3;              // 0..31 : A row in panel
  const int ak = (tid & 7) * 4;         // 0..28 : A k quad
  const int bk = tid >> 3;              // 0..31 : B k row
  const int bc = (tid & 7) * 8;         // 0..56 : B col oct
  const int a_row_g = (bm + ar < M) ? (bm + ar) : (M - 1);  // clamped (safe)
  const float* xrow = X + (size_t)a_row_g * K;

  // fragment source rows in LDS
  const int am  = (w >> 2) * 16 + l15;  // As row
  const int bnr = (w & 3) * 16 + l15;   // Bs row (= W column)

  auto stage = [&](int buf, int k0) {
#pragma unroll
    for (int i = 0; i < 4; ++i) {
      const int kk = k0 + ak + i;
      const int kc = (kk < K) ? kk : (K - 1);
      const float v = xrow[kc];                    // unconditional load
      As[buf][ar][ak + i] = (_Float16)((kk < K) ? v : 0.0f);
    }
    const int kk = k0 + bk;
    const int kc = (kk < K) ? kk : (K - 1);
    const float* wrow = W + (size_t)kc * N;
    const float kz = (kk < K) ? 1.0f : 0.0f;
#pragma unroll
    for (int i = 0; i < 8; ++i) {
      const int c  = bn + bc + i;
      const int cc = (c < N) ? c : (N - 1);
      Bs[buf][bc + i][bk] = (_Float16)(wrow[cc] * kz);  // unconditional load
    }
  };

  const int nchunks = (K + GEMM_BK - 1) / GEMM_BK;
  stage(0, 0);
  __syncthreads();

  v8f acc = {};
  for (int ch = 0; ch < nchunks; ++ch) {
    const int buf = ch & 1;
    if (ch + 1 < nchunks) stage(buf ^ 1, (ch + 1) * GEMM_BK);  // prefetch next

    const v8h alo = *(const v8h*)&As[buf][am][g * 8];
    const v8h ahi = *(const v8h*)&As[buf][am][16 + g * 8];
    const v8h blo = *(const v8h*)&Bs[buf][bnr][g * 8];
    const v8h bhi = *(const v8h*)&Bs[buf][bnr][16 + g * 8];
    v16h av, bv;
#pragma unroll
    for (int e = 0; e < 8; ++e) {
      av[e] = alo[e]; av[8 + e] = ahi[e];
      bv[e] = blo[e]; bv[8 + e] = bhi[e];
    }
    acc = __builtin_amdgcn_wmma_f32_16x16x32_f16(
        /*neg_a=*/false, av, /*neg_b=*/false, bv,
        /*c_mod=*/(short)0, acc, /*reuse_a=*/false, /*reuse_b=*/false);
    __syncthreads();   // readers of buf done; writers of buf^1 done
  }

  // ---- epilogue ----
  const int c     = tn + l15;
  const int rbase = tm + (g << 3);
  if (rbase + 8 <= M && c < N) {        // fast path (all M here are %16==0)
    float* yp = Y + (size_t)rbase * N + c;
    if (res) {
      const float* rp = res + (size_t)rbase * N + c;
      if (relu) {
#pragma unroll
        for (int j = 0; j < 8; ++j) yp[(size_t)j * N] = fmaxf(acc[j] + rp[(size_t)j * N], 0.0f);
      } else {
#pragma unroll
        for (int j = 0; j < 8; ++j) yp[(size_t)j * N] = acc[j] + rp[(size_t)j * N];
      }
    } else {
      if (relu) {
#pragma unroll
        for (int j = 0; j < 8; ++j) yp[(size_t)j * N] = fmaxf(acc[j], 0.0f);
      } else {
#pragma unroll
        for (int j = 0; j < 8; ++j) yp[(size_t)j * N] = acc[j];
      }
    }
  } else if (c < N) {                   // ragged-M slow path (defensive)
#pragma unroll
    for (int j = 0; j < 8; ++j) {
      const int r = rbase + j;
      if (r < M) {
        float v = acc[j];
        if (res)  v += res[(size_t)r * N + c];
        if (relu) v = fmaxf(v, 0.0f);
        Y[(size_t)r * N + c] = v;
      }
    }
  }
}

// ---------------------------------------------------------------------------
// Host-side orchestration helpers
// ---------------------------------------------------------------------------
static void launch_gemm(const float* X, const float* W, const float* R, float* Y,
                        int M, int K, int N, int relu, hipStream_t s) {
  dim3 grid(ceil_div(N, GEMM_BN), ceil_div(M, GEMM_BM));
  gemm_wmma_k<<<grid, TPB, 0, s>>>(X, W, R, Y, M, K, N, relu);
}

// One set-abstraction stage: ball query -> grouped features -> 3-layer MLP -> maxpool
static void sa_stage(const float* p_in, const float* x_in, const float* q,
                     int Nin, int Cin, int M, float radius,
                     const float* W0, int c0, const float* W1, int c1,
                     const float* W2, int c2,
                     int* bidx, float* bufA, float* bufB, float* x_out,
                     hipStream_t s) {
  const int B = 8, k = 32;
  const int totalM = B * M;
  const int rows = totalM * k;
  ball_group_k<<<ceil_div(totalM, TPB), TPB, 0, s>>>(p_in, q, bidx, Nin, M,
                                                     radius * radius, k, totalM);
  build_h_k<<<ceil_div(rows, TPB), TPB, 0, s>>>(p_in, x_in, q, bidx, bufA,
                                                Nin, M, k, Cin, rows);
  launch_gemm(bufA, W0, nullptr, bufB, rows, 3 + Cin, c0, 1, s);
  launch_gemm(bufB, W1, nullptr, bufA, rows, c0,      c1, 1, s);
  launch_gemm(bufA, W2, nullptr, bufB, rows, c1,      c2, 1, s);
  maxpool_k<<<ceil_div(totalM * c2, TPB), TPB, 0, s>>>(bufB, x_out, totalM, k, c2,
                                                       totalM * c2);
}

// One feature-propagation decoder (fp?1..fp?4): concat + interp + 2-layer MLPs
static void run_decode(const float* p0, const float* p1, const float* p2,
                       const float* p3, const float* p4,
                       const float* x1, const float* x2, const float* x3,
                       const float* x4, const float* const Wt[8],
                       float* cc3, float* y3, float* cc2, float* y2,
                       float* cc1, float* y1, float* cc0,
                       float* tmp, float* out, hipStream_t s) {
  // level 3: [x3(256) | interp(p3<-p4, x4)(512)] -> 768 -> 256 -> 256
  copy_strided_k<<<ceil_div(512 * 256, TPB), TPB, 0, s>>>(x3, cc3, 256, 768, 0,
                                                          512 * 256);
  interp3_k<<<ceil_div(8 * 64, TPB), TPB, 0, s>>>(p3, p4, x4, cc3, 64, 16, 512,
                                                  768, 256, 8 * 64);
  launch_gemm(cc3, Wt[0], nullptr, tmp, 512, 768, 256, 1, s);
  launch_gemm(tmp, Wt[1], nullptr, y3, 512, 256, 256, 1, s);
  // level 2: [x2(128) | interp(p2<-p3, y3)(256)] -> 384 -> 256 -> 256
  copy_strided_k<<<ceil_div(2048 * 128, TPB), TPB, 0, s>>>(x2, cc2, 128, 384, 0,
                                                           2048 * 128);
  interp3_k<<<ceil_div(2048, TPB), TPB, 0, s>>>(p2, p3, y3, cc2, 256, 64, 256,
                                                384, 128, 2048);
  launch_gemm(cc2, Wt[2], nullptr, tmp, 2048, 384, 256, 1, s);
  launch_gemm(tmp, Wt[3], nullptr, y2, 2048, 256, 256, 1, s);
  // level 1: [x1(64) | interp(p1<-p2, y2)(256)] -> 320 -> 256 -> 128
  copy_strided_k<<<ceil_div(8192 * 64, TPB), TPB, 0, s>>>(x1, cc1, 64, 320, 0,
                                                          8192 * 64);
  interp3_k<<<ceil_div(8192, TPB), TPB, 0, s>>>(p1, p2, y2, cc1, 1024, 256, 256,
                                                320, 64, 8192);
  launch_gemm(cc1, Wt[4], nullptr, tmp, 8192, 320, 256, 1, s);
  launch_gemm(tmp, Wt[5], nullptr, y1, 8192, 256, 128, 1, s);
  // level 0: interp(p0<-p1, y1)(128) -> 128 -> 128
  interp3_k<<<ceil_div(32768, TPB), TPB, 0, s>>>(p0, p1, y1, cc0, 4096, 1024, 128,
                                                 128, 0, 32768);
  launch_gemm(cc0, Wt[6], nullptr, tmp, 32768, 128, 128, 1, s);
  launch_gemm(tmp, Wt[7], nullptr, out, 32768, 128, 128, 1, s);
}

// ---------------------------------------------------------------------------
// Parameter table: 33 weight arrays.
// Canonical (dict insertion) order: sa1(3) sa2(3) sa3(3) sa4(3)
//   fp11(2) fp12(2) fp13(2) fp14(2) fp21(2) fp22(2) fp23(2) fp24(2)
//   adapt sem_h sem_out ins_h ins_out
// If the harness passes a single concatenated params blob (n_in==3), leaves are
// assumed in jax tree-flatten (sorted-key) order; map back via tables below.
// ---------------------------------------------------------------------------
static const int kSortedToCanon[33] = {
  28, 12, 13, 14, 15, 16, 17, 18, 19, 20, 21, 22, 23, 24, 25, 26, 27,
  31, 32, 0, 1, 2, 3, 4, 5, 6, 7, 8, 9, 10, 11, 29, 30 };
static const size_t kSortedSizes[33] = {
  16384,                                  // adapt
  196608, 65536, 98304, 65536, 81920, 32768, 16384, 16384,   // fp11..fp14
  196608, 65536, 98304, 65536, 81920, 32768, 16384, 16384,   // fp21..fp24
  16384, 640,                             // ins_h, ins_out
  384, 1024, 2048, 4288, 4096, 8192,      // sa1, sa2
  16768, 16384, 32768, 66304, 65536, 131072, // sa3, sa4
  16384, 1664 };                          // sem_h, sem_out

extern "C" void kernel_launch(void* const* d_in, const int* in_sizes, int n_in,
                              void* d_out, int out_size, void* d_ws, size_t ws_size,
                              hipStream_t stream) {
  (void)in_sizes; (void)out_size; (void)ws_size;

  const float* p0 = (const float*)d_in[0];   // [8,4096,3]
  const float* x0 = (const float*)d_in[1];   // [8,4096,9]

  const float* P[33];
  if (n_in >= 35) {
    for (int i = 0; i < 33; ++i) P[i] = (const float*)d_in[2 + i];
  } else {
    const float* base = (const float*)d_in[2];
    size_t poff = 0;
    for (int i = 0; i < 33; ++i) { P[kSortedToCanon[i]] = base + poff; poff += kSortedSizes[i]; }
  }

  // ---- workspace bump allocator ----
  size_t off = 0;
  char* wsp = (char*)d_ws;
  auto falloc = [&](size_t elems) -> float* {
    float* p = (float*)(wsp + off);
    off += (elems * sizeof(float) + 255) & ~(size_t)255;
    return p;
  };
  auto ialloc = [&](size_t elems) -> int* {
    int* p = (int*)(wsp + off);
    off += (elems * sizeof(int) + 255) & ~(size_t)255;
    return p;
  };

  int* idx1 = ialloc(8 * 1024);
  int* idx2 = ialloc(8 * 256);
  int* idx3 = ialloc(8 * 64);
  int* idx4 = ialloc(8 * 16);
  int* bidx = ialloc((size_t)8 * 1024 * 32);
  float* p1 = falloc(8 * 1024 * 3);
  float* p2 = falloc(8 * 256 * 3);
  float* p3 = falloc(8 * 64 * 3);
  float* p4 = falloc(8 * 16 * 3);
  float* bufA = falloc((size_t)16777216);          // ping
  float* bufB = falloc((size_t)16777216);          // pong (also decode tmp)
  float* x1 = falloc((size_t)8 * 1024 * 64);
  float* x2 = falloc((size_t)8 * 256 * 128);
  float* x3 = falloc((size_t)8 * 64 * 256);
  float* x4 = falloc((size_t)8 * 16 * 512);
  float* fsem = falloc((size_t)32768 * 128);
  float* fins = falloc((size_t)32768 * 128);
  float* cc3 = falloc((size_t)512 * 768);
  float* y3  = falloc((size_t)512 * 256);
  float* cc2 = falloc((size_t)2048 * 384);
  float* y2  = falloc((size_t)2048 * 256);
  float* cc1 = falloc((size_t)8192 * 320);
  float* y1  = falloc((size_t)8192 * 128);
  float* cc0 = falloc((size_t)32768 * 128);

  // ---- encoder: FPS + set abstraction pyramid ----
  fps_kernel<<<8, TPB, 0, stream>>>(p0, idx1, 4096, 1024);
  gather_pts_k<<<ceil_div(8 * 1024, TPB), TPB, 0, stream>>>(p0, idx1, p1, 4096, 1024, 8 * 1024);
  sa_stage(p0, x0, p1, 4096, 9, 1024, 0.1f,
           P[0], 32, P[1], 32, P[2], 64, bidx, bufA, bufB, x1, stream);

  fps_kernel<<<8, TPB, 0, stream>>>(p1, idx2, 1024, 256);
  gather_pts_k<<<ceil_div(8 * 256, TPB), TPB, 0, stream>>>(p1, idx2, p2, 1024, 256, 8 * 256);
  sa_stage(p1, x1, p2, 1024, 64, 256, 0.2f,
           P[3], 64, P[4], 64, P[5], 128, bidx, bufA, bufB, x2, stream);

  fps_kernel<<<8, TPB, 0, stream>>>(p2, idx3, 256, 64);
  gather_pts_k<<<ceil_div(8 * 64, TPB), TPB, 0, stream>>>(p2, idx3, p3, 256, 64, 8 * 64);
  sa_stage(p2, x2, p3, 256, 128, 64, 0.4f,
           P[6], 128, P[7], 128, P[8], 256, bidx, bufA, bufB, x3, stream);

  fps_kernel<<<8, TPB, 0, stream>>>(p3, idx4, 64, 16);
  gather_pts_k<<<ceil_div(8 * 16, TPB), TPB, 0, stream>>>(p3, idx4, p4, 64, 16, 8 * 16);
  sa_stage(p3, x3, p4, 64, 256, 16, 0.8f,
           P[9], 256, P[10], 256, P[11], 512, bidx, bufA, bufB, x4, stream);

  // ---- decoders ----
  const float* Wfp1[8] = { P[12], P[13], P[14], P[15], P[16], P[17], P[18], P[19] };
  const float* Wfp2[8] = { P[20], P[21], P[22], P[23], P[24], P[25], P[26], P[27] };
  run_decode(p0, p1, p2, p3, p4, x1, x2, x3, x4, Wfp1,
             cc3, y3, cc2, y2, cc1, y1, cc0, bufB, fsem, stream);
  run_decode(p0, p1, p2, p3, p4, x1, x2, x3, x4, Wfp2,
             cc3, y3, cc2, y2, cc1, y1, cc0, bufB, fins, stream);

  // fins = decode_fp2 + fsem @ adapt   (per-element in-place residual is safe)
  launch_gemm(fsem, P[28], fins, fins, 32768, 128, 128, 0, stream);

  // ---- heads ----
  float* outp = (float*)d_out;                     // [sem: 32768*13][emb: 32768*5]
  launch_gemm(fsem, P[29], nullptr, bufA, 32768, 128, 128, 1, stream);
  launch_gemm(bufA, P[30], nullptr, outp, 32768, 128, 13, 0, stream);
  launch_gemm(fins, P[31], nullptr, bufA, 32768, 128, 128, 1, stream);
  launch_gemm(bufA, P[32], nullptr, outp + (size_t)32768 * 13, 32768, 128, 5, 0, stream);
}
